// NONLocalBlock2D_44057774522562
// MI455X (gfx1250) — matrix operand order repaired and checked by reading, not compiled, not run on hardware
//
#include <hip/hip_runtime.h>
#include <hip/hip_bf16.h>

// ---------------- problem constants ----------------
constexpr int B_  = 8;
constexpr int C_  = 512;   // channels
constexpr int CI_ = 256;   // inter channels
constexpr int H_  = 64, W_ = 64;
constexpr int HW_ = H_ * W_;       // 4096 queries
constexpr int NK_ = (H_/2)*(W_/2); // 1024 keys

typedef __attribute__((ext_vector_type(16))) __bf16 bf16x16;
typedef __attribute__((ext_vector_type(8)))  __bf16 bf16x8;
typedef __attribute__((ext_vector_type(8)))  float  f32x8;

// ---------------- bf16 helper: native converting cast (v_cvt on gfx1250) ----------------
__device__ __forceinline__ __bf16 f2bf(float f) { return (__bf16)f; }

// ---------------- WMMA fragment loads (vectorized from LDS) ----------------
// A (16x32 MxK), LDS tile row-major [M][stride], stride multiple of 8.
// lane m = lane&15; koff = 8*(lane>=16). Elements e0..7 -> K=koff+e,
// e8..15 -> K=16+koff+(e-8): two contiguous 16B runs -> two ds_load_b128.
__device__ __forceinline__ bf16x16 load_a_frag(const __bf16* lds, int stride) {
  const int lane = threadIdx.x & 31;
  const int m    = lane & 15;
  const int koff = (lane >> 4) << 3;
  const __bf16* p = lds + m * stride + koff;
  const bf16x8 lo = *(const bf16x8*)(p);
  const bf16x8 hi = *(const bf16x8*)(p + 16);
  return __builtin_shufflevector(lo, hi, 0,1,2,3,4,5,6,7,8,9,10,11,12,13,14,15);
}

// B (32x16 KxN) staged TRANSPOSED in LDS as [N][stride]; kbase = 8*(lane>=16):
// e0..7 -> K=kbase+e, e8..15 -> K=16+kbase+(e-8) -> two ds_load_b128.
__device__ __forceinline__ bf16x16 load_bT_frag(const __bf16* ldsT, int stride) {
  const int lane  = threadIdx.x & 31;
  const int n     = lane & 15;
  const int kbase = (lane >> 4) << 3;
  const __bf16* p = ldsT + n * stride + kbase;
  const bf16x8 lo = *(const bf16x8*)(p);
  const bf16x8 hi = *(const bf16x8*)(p + 16);
  return __builtin_shufflevector(lo, hi, 0,1,2,3,4,5,6,7,8,9,10,11,12,13,14,15);
}

// C/D f32 16x16: lane = N (mod 16), VGPR r -> row M = 8*(lane>=16) + r.

// ---------------- K0: convert weights to bf16 ----------------
__global__ void k_cvt_weights(const float* __restrict__ gw, const float* __restrict__ Ww,
                              __bf16* __restrict__ gw_b, __bf16* __restrict__ Ww_b) {
  const int i = blockIdx.x * blockDim.x + threadIdx.x;
  if (i < CI_ * C_) { gw_b[i] = f2bf(gw[i]); Ww_b[i] = f2bf(Ww[i]); }
}

// ---------------- K1: phi[b][k][c] = bf16(maxpool2(x)) (k-major for B-staging) --------
__global__ void k_phi_pool(const float* __restrict__ x, __bf16* __restrict__ phi) {
  const int i = blockIdx.x * blockDim.x + threadIdx.x;
  if (i >= B_ * NK_ * C_) return;
  const int c = i % C_;
  const int k = (i / C_) % NK_;
  const int b = i / (C_ * NK_);
  const int hp = k >> 5, wp = k & 31;
  const float* p = x + ((size_t)(b * C_ + c)) * HW_ + (hp * 2) * W_ + wp * 2;
  const float m = fmaxf(fmaxf(p[0], p[1]), fmaxf(p[W_], p[W_ + 1]));
  phi[i] = f2bf(m);
}

// ---------------- K2: gfull[ci][n] = g_w @ x + g_b  (WMMA GEMM) ----------------
__global__ __launch_bounds__(128)
void k_g_gemm(const float* __restrict__ x, const __bf16* __restrict__ gw,
              const float* __restrict__ gbias, float* __restrict__ gfull) {
  __shared__ __attribute__((aligned(16))) __bf16 As[64][40];   // [m][k]
  __shared__ __attribute__((aligned(16))) __bf16 Bs[64][40];   // transposed: [n][k]
  const int tid = threadIdx.x, wv = tid >> 5, lane = tid & 31;
  const int mb = (lane >> 4) << 3, nn = lane & 15;
  const int b  = blockIdx.z;
  const int m0 = blockIdx.y * 64;
  const int n0 = blockIdx.x * 64;

  const f32x8 zero = {};
  f32x8 acc[4];
#pragma unroll
  for (int t = 0; t < 4; ++t) acc[t] = zero;

  for (int ks = 0; ks < C_ / 32; ++ks) {
    const int kk0 = ks * 32;
    __syncthreads();
#pragma unroll
    for (int it = 0; it < 2; ++it) {                           // A: 16B copies
      const int idx = tid + it * 128;
      const int m = idx >> 2, kb = (idx & 3) * 8;
      *(bf16x8*)&As[m][kb] = *(const bf16x8*)&gw[(size_t)(m0 + m) * C_ + kk0 + kb];
    }
#pragma unroll
    for (int it = 0; it < 4; ++it) {                           // B: f32x4 + transpose
      const int idx = tid + it * 128;
      const int k = idx >> 4, nb = (idx & 15) * 4;
      const float4 vv = *(const float4*)&x[((size_t)(b * C_ + kk0 + k)) * HW_ + n0 + nb];
      Bs[nb + 0][k] = f2bf(vv.x); Bs[nb + 1][k] = f2bf(vv.y);
      Bs[nb + 2][k] = f2bf(vv.z); Bs[nb + 3][k] = f2bf(vv.w);
    }
    __syncthreads();
    const bf16x16 a = load_a_frag(&As[wv * 16][0], 40);
#pragma unroll
    for (int t = 0; t < 4; ++t) {
      const bf16x16 bb = load_bT_frag(&Bs[t * 16][0], 40);
      acc[t] = __builtin_amdgcn_wmma_f32_16x16x32_bf16(false, a, false, bb,
                                                       (short)0, acc[t], false, false);
    }
  }
#pragma unroll
  for (int r = 0; r < 8; ++r) {
    const int ci = m0 + wv * 16 + mb + r;
    const float bias = gbias[ci];
#pragma unroll
    for (int t = 0; t < 4; ++t)
      gfull[((size_t)(b * CI_ + ci)) * HW_ + n0 + t * 16 + nn] = acc[t][r] + bias;
  }
}

// ---------------- K3: gkv[b][ci][k] = bf16(maxpool2(gfull)) (ci-major) ----------------
__global__ void k_g_pool(const float* __restrict__ gfull, __bf16* __restrict__ gkv) {
  const int i = blockIdx.x * blockDim.x + threadIdx.x;
  if (i >= B_ * CI_ * NK_) return;
  const int k  = i % NK_;
  const int ci = (i / NK_) % CI_;
  const int b  = i / (NK_ * CI_);
  const int hp = k >> 5, wp = k & 31;
  const float* p = gfull + ((size_t)(b * CI_ + ci)) * HW_ + (hp * 2) * W_ + wp * 2;
  const float m = fmaxf(fmaxf(p[0], p[1]), fmaxf(p[W_], p[W_ + 1]));
  gkv[i] = f2bf(m);
}

// ---------------- K4: flash attention, y[b][q][ci] ----------------
__global__ __launch_bounds__(128)
void k_attention(const float* __restrict__ x, const __bf16* __restrict__ phi,
                 const __bf16* __restrict__ gkv, __bf16* __restrict__ y) {
  __shared__ __attribute__((aligned(16))) __bf16 Qs[64][40];    // [q][c]   (A tiles)
  __shared__ __attribute__((aligned(16))) __bf16 Ks[32][40];    // [key][c] (B^T tiles)
  __shared__ __attribute__((aligned(16))) __bf16 Gs[256][40];   // [ci][k]  (B^T tiles)
  __shared__ __attribute__((aligned(16))) __bf16 Ps[4][16][40]; // per-wave P (A layout)

  const int tid = threadIdx.x, wv = tid >> 5, lane = tid & 31;
  const int mb = (lane >> 4) << 3, nn = lane & 15;
  const int b  = blockIdx.y;
  const int q0 = blockIdx.x * 64;

  const f32x8 zero = {};
  f32x8 O[16];
#pragma unroll
  for (int t = 0; t < 16; ++t) O[t] = zero;
  float m_run[8], l_run[8];
#pragma unroll
  for (int r = 0; r < 8; ++r) { m_run[r] = -1e30f; l_run[r] = 0.f; }

  for (int kt = 0; kt < NK_ / 32; ++kt) {
    const int k0 = kt * 32;
    __syncthreads();                          // all waves done with prev Gs / Ps
    // stage V tile (bf16, both sides contiguous 16B chunks)
#pragma unroll
    for (int it = 0; it < 8; ++it) {
      const int idx = tid + it * 128;
      const int ci = idx >> 2, kb = (idx & 3) * 8;
      const __bf16* src = &gkv[((size_t)(b * CI_ + ci)) * NK_ + k0 + kb];
      *(bf16x8*)&Gs[ci][kb] = *(const bf16x8*)src;
      if (kt + 1 < NK_ / 32)
        __builtin_prefetch(src + 32, 0, 0);
    }

    f32x8 s0 = zero, s1 = zero;
    for (int cs = 0; cs < C_ / 32; ++cs) {
      const int c0 = cs * 32;
      __syncthreads();
#pragma unroll
      for (int it = 0; it < 4; ++it) {                        // Q: f32x4 + transpose
        const int idx = tid + it * 128;
        const int cc = idx >> 4, qb = (idx & 15) * 4;
        const float4 vv = *(const float4*)&x[((size_t)(b * C_ + c0 + cc)) * HW_ + q0 + qb];
        Qs[qb + 0][cc] = f2bf(vv.x); Qs[qb + 1][cc] = f2bf(vv.y);
        Qs[qb + 2][cc] = f2bf(vv.z); Qs[qb + 3][cc] = f2bf(vv.w);
      }
      {                                                       // K: 16B copies (1/thread)
        const int kk = tid >> 2, cb = (tid & 3) * 8;
        *(bf16x8*)&Ks[kk][cb] =
            *(const bf16x8*)&phi[((size_t)(b * NK_ + k0 + kk)) * C_ + c0 + cb];
      }
      __syncthreads();
      const bf16x16 a  = load_a_frag(&Qs[wv * 16][0], 40);
      const bf16x16 b0 = load_bT_frag(&Ks[0][0], 40);
      const bf16x16 b1 = load_bT_frag(&Ks[16][0], 40);
      s0 = __builtin_amdgcn_wmma_f32_16x16x32_bf16(false, a, false, b0, (short)0, s0, false, false);
      s1 = __builtin_amdgcn_wmma_f32_16x16x32_bf16(false, a, false, b1, (short)0, s1, false, false);
    }

    // online softmax over this 32-key tile (rows live in half-waves of 16 lanes)
#pragma unroll
    for (int r = 0; r < 8; ++r) {
      float v = fmaxf(s0[r], s1[r]);
#pragma unroll
      for (int off = 1; off < 16; off <<= 1) v = fmaxf(v, __shfl_xor(v, off));
      const float mnew  = fmaxf(m_run[r], v);
      const float scale = __expf(m_run[r] - mnew);
      const float p0 = __expf(s0[r] - mnew);
      const float p1 = __expf(s1[r] - mnew);
      float ps = p0 + p1;
#pragma unroll
      for (int off = 1; off < 16; off <<= 1) ps += __shfl_xor(ps, off);
      l_run[r] = l_run[r] * scale + ps;
      m_run[r] = mnew;
#pragma unroll
      for (int t = 0; t < 16; ++t) O[t][r] *= scale;
      Ps[wv][mb + r][nn]      = f2bf(p0);
      Ps[wv][mb + r][16 + nn] = f2bf(p1);
    }
    __syncthreads();

    const bf16x16 pa = load_a_frag(&Ps[wv][0][0], 40);
#pragma unroll
    for (int t = 0; t < 16; ++t) {
      const bf16x16 gb = load_bT_frag(&Gs[t * 16][0], 40);
      O[t] = __builtin_amdgcn_wmma_f32_16x16x32_bf16(false, pa, false, gb,
                                                     (short)0, O[t], false, false);
    }
  }

#pragma unroll
  for (int r = 0; r < 8; ++r) {
    const float inv = 1.0f / l_run[r];
    const int q = q0 + wv * 16 + mb + r;
#pragma unroll
    for (int t = 0; t < 16; ++t)
      y[((size_t)(b * HW_ + q)) * CI_ + t * 16 + nn] = f2bf(O[t][r] * inv);
  }
}

// ---------------- K5: out = BN(W_w @ y + W_b) + x ----------------
__global__ __launch_bounds__(128)
void k_out_gemm(const __bf16* __restrict__ Wwb, const __bf16* __restrict__ y,
                const float* __restrict__ Wb, const float* __restrict__ gamma,
                const float* __restrict__ beta, const float* __restrict__ mean,
                const float* __restrict__ var, const float* __restrict__ x,
                float* __restrict__ out) {
  __shared__ __attribute__((aligned(16))) __bf16 As[64][40];   // [c][ci]
  __shared__ __attribute__((aligned(16))) __bf16 Bs[64][40];   // transposed: [n][ci]
  const int tid = threadIdx.x, wv = tid >> 5, lane = tid & 31;
  const int mb = (lane >> 4) << 3, nn = lane & 15;
  const int b  = blockIdx.z;
  const int m0 = blockIdx.y * 64;
  const int n0 = blockIdx.x * 64;

  const f32x8 zero = {};
  f32x8 acc[4];
#pragma unroll
  for (int t = 0; t < 4; ++t) acc[t] = zero;

  for (int ks = 0; ks < CI_ / 32; ++ks) {
    const int kk0 = ks * 32;
    __syncthreads();
#pragma unroll
    for (int it = 0; it < 2; ++it) {                           // A: 16B copies
      const int idx = tid + it * 128;
      const int m = idx >> 2, kb = (idx & 3) * 8;
      *(bf16x8*)&As[m][kb] = *(const bf16x8*)&Wwb[(size_t)(m0 + m) * CI_ + kk0 + kb];
    }
#pragma unroll
    for (int it = 0; it < 2; ++it) {                           // B^T: 16B copies
      const int idx = tid + it * 128;
      const int n = idx >> 2, kb = (idx & 3) * 8;
      *(bf16x8*)&Bs[n][kb] =
          *(const bf16x8*)&y[((size_t)(b * HW_ + n0 + n)) * CI_ + kk0 + kb];
    }
    __syncthreads();
    const bf16x16 a = load_a_frag(&As[wv * 16][0], 40);
#pragma unroll
    for (int t = 0; t < 4; ++t) {
      const bf16x16 bb = load_bT_frag(&Bs[t * 16][0], 40);
      acc[t] = __builtin_amdgcn_wmma_f32_16x16x32_bf16(false, a, false, bb,
                                                       (short)0, acc[t], false, false);
    }
  }
#pragma unroll
  for (int r = 0; r < 8; ++r) {
    const int c = m0 + wv * 16 + mb + r;
    const float sc = gamma[c] * rsqrtf(var[c] + 1e-5f);
    const float sh = (Wb[c] - mean[c]) * sc + beta[c];
#pragma unroll
    for (int t = 0; t < 4; ++t) {
      const size_t o = ((size_t)(b * C_ + c)) * HW_ + n0 + t * 16 + nn;
      out[o] = acc[t][r] * sc + sh + x[o];
    }
  }
}

// ---------------- launch ----------------
extern "C" void kernel_launch(void* const* d_in, const int* in_sizes, int n_in,
                              void* d_out, int out_size, void* d_ws, size_t ws_size,
                              hipStream_t stream) {
  (void)in_sizes; (void)n_in; (void)out_size; (void)ws_size;
  const float* x        = (const float*)d_in[0];
  const float* g_w      = (const float*)d_in[1];
  const float* g_b      = (const float*)d_in[2];
  const float* W_w      = (const float*)d_in[3];
  const float* W_b      = (const float*)d_in[4];
  const float* bn_gamma = (const float*)d_in[5];
  const float* bn_beta  = (const float*)d_in[6];
  const float* bn_mean  = (const float*)d_in[7];
  const float* bn_var   = (const float*)d_in[8];
  float* out = (float*)d_out;

  char* ws = (char*)d_ws;
  size_t off = 0;
  auto take = [&](size_t bytes) -> char* {
    char* p = ws + off;
    off = (off + bytes + 255) & ~(size_t)255;
    return p;
  };
  __bf16* gw_b  = (__bf16*)take((size_t)CI_ * C_ * 2);
  __bf16* Ww_b  = (__bf16*)take((size_t)C_ * CI_ * 2);
  __bf16* phi   = (__bf16*)take((size_t)B_ * NK_ * C_ * 2);   // [b][k][c]
  float*  gfull = (float*) take((size_t)B_ * CI_ * HW_ * 4);  // [b][ci][n]
  __bf16* gkv   = (__bf16*)take((size_t)B_ * CI_ * NK_ * 2);  // [b][ci][k]
  __bf16* ybuf  = (__bf16*)take((size_t)B_ * HW_ * CI_ * 2);  // [b][q][ci]

  k_cvt_weights<<<(CI_ * C_ + 255) / 256, 256, 0, stream>>>(g_w, W_w, gw_b, Ww_b);
  k_phi_pool<<<(B_ * NK_ * C_ + 255) / 256, 256, 0, stream>>>(x, phi);
  k_g_gemm<<<dim3(HW_ / 64, CI_ / 64, B_), 128, 0, stream>>>(x, gw_b, g_b, gfull);
  k_g_pool<<<(B_ * CI_ * NK_ + 255) / 256, 256, 0, stream>>>(gfull, gkv);
  k_attention<<<dim3(HW_ / 64, B_), 128, 0, stream>>>(x, phi, gkv, ybuf);
  k_out_gemm<<<dim3(HW_ / 64, C_ / 64, B_), 128, 0, stream>>>(Ww_b, ybuf, W_b, bn_gamma,
                                                              bn_beta, bn_mean, bn_var, x, out);
}
